// MultiQueryAttention_23072564314373
// MI455X (gfx1250) — matrix operand order repaired and checked by reading, not compiled, not run on hardware
//
#include <hip/hip_runtime.h>

typedef __attribute__((ext_vector_type(16))) _Float16 v16h;
typedef __attribute__((ext_vector_type(8)))  _Float16 v8h;
typedef __attribute__((ext_vector_type(8)))  float    v8f;

#define BATCH 2
#define SEQ   2048
#define DM    2048
#define NH    16
#define HDIM  128

union F16Frag { v16h v; v8h h[2]; };

static __device__ __forceinline__ v8f vzero8() {
  v8f z;
#pragma unroll
  for (int i = 0; i < 8; ++i) z[i] = 0.0f;
  return z;
}

// Async copy of one 16B chunk from global to LDS (per-lane), ASYNCcnt-tracked.
// saddr form: vdst = LDS byte address, vaddr = 32-bit byte offset, saddr = base.
static __device__ __forceinline__ void async_b128(uint32_t lds_addr,
                                                  const void* sbase,
                                                  uint32_t voff_bytes) {
  asm volatile("global_load_async_to_lds_b128 %0, %1, %2"
               :: "v"(lds_addr), "v"(voff_bytes), "s"(sbase)
               : "memory");
}

// ---------------------------------------------------------------------------
// Elementwise f32 -> f16 (n must be a multiple of 2048; true for all uses)
// ---------------------------------------------------------------------------
__global__ __launch_bounds__(256)
void cvt_f16(const float* __restrict__ in, _Float16* __restrict__ out, int n) {
  int i0 = (blockIdx.x * 256 + threadIdx.x) * 8;
  if (i0 >= n) return;
  float4 a = *(const float4*)&in[i0];
  float4 b = *(const float4*)&in[i0 + 4];
  v8h o;
  o[0] = (_Float16)a.x; o[1] = (_Float16)a.y; o[2] = (_Float16)a.z; o[3] = (_Float16)a.w;
  o[4] = (_Float16)b.x; o[5] = (_Float16)b.y; o[6] = (_Float16)b.z; o[7] = (_Float16)b.w;
  *(v8h*)&out[i0] = o;
}

// ---------------------------------------------------------------------------
// W[K,N] f32 -> Wt[N,K] f16, 64x64 LDS tile transpose. Grid (N/64, K/64).
// ---------------------------------------------------------------------------
__global__ __launch_bounds__(256)
void transpose_cvt_f16(const float* __restrict__ W, _Float16* __restrict__ Wt,
                       int K, int N) {
  __shared__ _Float16 tile[64][65];
  const int k0 = blockIdx.y * 64, n0 = blockIdx.x * 64;
  const int t = threadIdx.x;
#pragma unroll
  for (int i = 0; i < 16; ++i) {
    int e = t + 256 * i;
    int r = e >> 6, c = e & 63;
    tile[c][r] = (_Float16)W[(size_t)(k0 + r) * N + n0 + c];
  }
  __syncthreads();
#pragma unroll
  for (int i = 0; i < 16; ++i) {
    int e = t + 256 * i;
    int r = e >> 6, c = e & 63;
    Wt[(size_t)(n0 + r) * K + k0 + c] = tile[r][c];
  }
}

// ---------------------------------------------------------------------------
// TN GEMM: C[M,N] = A[M,K] (f16) @ Wt[N,K]^T (f16) + bias, f32 accum.
// Tile 128x128, K-block 32, double-buffered async global->LDS staging.
// 8 waves: wave = (wm: 64-row slab) x (wn: 32-col slab).
// Requires M%128==0, (N%128==0 or N==128), K%32==0.
// ---------------------------------------------------------------------------
template <bool F16OUT>
__global__ __launch_bounds__(256)
void gemm_tn_f16(const _Float16* __restrict__ A, const _Float16* __restrict__ Wt,
                 const float* __restrict__ bias, void* __restrict__ Cout,
                 int M, int N, int K) {
  __shared__ alignas(16) _Float16 As[2][128][40];  // [buf][m][k] stride 80B
  __shared__ alignas(16) _Float16 Ws[2][128][40];  // [buf][n][k] stride 80B

  const int tid  = threadIdx.x;
  const int lane = tid & 31;
  const int w    = tid >> 5;
  const int hi   = (lane >= 16) ? 1 : 0;
  const int ln   = lane & 15;
  const int m0   = blockIdx.y * 128;
  const int n0   = blockIdx.x * 128;
  const int wm   = w & 1;    // 0..1 -> 64-row slab
  const int wn   = w >> 1;   // 0..3 -> 32-col slab

  // staging geometry: 128 rows x 32 halfs = 512 x 16B chunks, 2 per thread
  int crow[2], cko[2];
#pragma unroll
  for (int i = 0; i < 2; ++i) {
    int c = tid + 256 * i;
    crow[i] = c >> 2;
    cko[i]  = (c & 3) * 8;
  }
  const _Float16* Ab = A  + (size_t)m0 * K;
  const _Float16* Wb = Wt + (size_t)n0 * K;

  auto issue = [&](int buf, int k0) {
#pragma unroll
    for (int i = 0; i < 2; ++i) {
      uint32_t la = (uint32_t)(uintptr_t)&As[buf][crow[i]][cko[i]];
      async_b128(la, Ab, (uint32_t)((crow[i] * K + k0 + cko[i]) * 2));
      uint32_t lw = (uint32_t)(uintptr_t)&Ws[buf][crow[i]][cko[i]];
      async_b128(lw, Wb, (uint32_t)((crow[i] * K + k0 + cko[i]) * 2));
    }
  };

  v8f acc[4][2];
#pragma unroll
  for (int a = 0; a < 4; ++a)
#pragma unroll
    for (int b = 0; b < 2; ++b) acc[a][b] = vzero8();

  issue(0, 0);  // 4 async ops outstanding
  const int nkb = K / 32;
  for (int kb = 0; kb < nkb; ++kb) {
    const int buf = kb & 1;
    if (kb + 1 < nkb) {
      issue(buf ^ 1, (kb + 1) * 32);                       // 8 outstanding
      asm volatile("s_wait_asynccnt 0x4" ::: "memory");    // current buf landed
    } else {
      asm volatile("s_wait_asynccnt 0x0" ::: "memory");
    }
    __syncthreads();

    // B fragments: lane -> col (ln), element e -> K = e + 8*((e>=8)+hi)
    F16Frag bf[2];
#pragma unroll
    for (int tn = 0; tn < 2; ++tn) {
      int n = wn * 32 + tn * 16 + ln;
      bf[tn].h[0] = *(const v8h*)&Ws[buf][n][8 * hi];
      bf[tn].h[1] = *(const v8h*)&Ws[buf][n][8 * hi + 16];
    }
    // A fragments: lane -> row (ln), element e -> K = e + 8*((e>=8)+hi)
#pragma unroll
    for (int tm = 0; tm < 4; ++tm) {
      F16Frag af;
      int row = wm * 64 + tm * 16 + ln;
      af.h[0] = *(const v8h*)&As[buf][row][8 * hi];
      af.h[1] = *(const v8h*)&As[buf][row][8 * hi + 16];
#pragma unroll
      for (int tn = 0; tn < 2; ++tn) {
        acc[tm][tn] = __builtin_amdgcn_wmma_f32_16x16x32_f16(
            false, af.v, false, bf[tn].v, (short)0, acc[tm][tn], false, false);
      }
    }
    __syncthreads();
  }

  // epilogue: C layout -> VGPR r holds row r + 8*hi, col = ln (per 16x16 tile)
#pragma unroll
  for (int tm = 0; tm < 4; ++tm)
#pragma unroll
    for (int tn = 0; tn < 2; ++tn) {
      int col = n0 + wn * 32 + tn * 16 + ln;
      float bsc = bias[col];
#pragma unroll
      for (int r = 0; r < 8; ++r) {
        int row = m0 + wm * 64 + tm * 16 + r + 8 * hi;
        float val = acc[tm][tn][r] + bsc;
        if constexpr (F16OUT)
          ((_Float16*)Cout)[(size_t)row * N + col] = (_Float16)val;
        else
          ((float*)Cout)[(size_t)row * N + col] = val;
      }
    }
}

// ---------------------------------------------------------------------------
// Flash-style causal MQA core: per (b, h, 128-row query block).
// q: [B,S,D] f16 (head h at cols h*128..), k,v: [B,S,HD] f16, o: [B,S,D] f16.
// ---------------------------------------------------------------------------
__global__ __launch_bounds__(256)
void mqa_flash(const _Float16* __restrict__ q, const _Float16* __restrict__ kk,
               const _Float16* __restrict__ vv, _Float16* __restrict__ o) {
  __shared__ alignas(16) _Float16 Kt[64][136];      // [kv][hd]  stride 272B
  __shared__ alignas(16) _Float16 Vt[128][72];      // [hd][kv]  stride 144B
  __shared__ alignas(16) _Float16 Pst[8][16][72];   // per-wave P staging

  const int tid  = threadIdx.x;
  const int lane = tid & 31;
  const int w    = tid >> 5;
  const int hi   = (lane >= 16) ? 1 : 0;
  const int ln   = lane & 15;
  const int m0   = blockIdx.x * 128;
  const int h    = blockIdx.y;
  const int b    = blockIdx.z;
  const float scale = 0.08838834764831845f;  // 1/sqrt(128)

  F16Frag qf[4];
  {
    int row = m0 + w * 16 + ln;
    size_t base = ((size_t)b * SEQ + row) * DM + (size_t)h * HDIM;
#pragma unroll
    for (int c = 0; c < 4; ++c) {
      int kb = c * 32 + 8 * hi;
      qf[c].h[0] = *(const v8h*)&q[base + kb];
      qf[c].h[1] = *(const v8h*)&q[base + kb + 16];
    }
  }

  v8f oacc[8];
#pragma unroll
  for (int i = 0; i < 8; ++i) oacc[i] = vzero8();
  float mrow[8], lrow[8];
#pragma unroll
  for (int r = 0; r < 8; ++r) { mrow[r] = -__builtin_inff(); lrow[r] = 0.0f; }

  const int jmax = (m0 + 127) >> 6;
  for (int j = 0; j <= jmax; ++j) {
#pragma unroll
    for (int i = 0; i < 4; ++i) {
      int c = tid + 256 * i;
      int kvr = c >> 4, ho = (c & 15) * 8;
      size_t gb = ((size_t)b * SEQ + j * 64 + kvr) * HDIM + ho;
      v8h k8 = *(const v8h*)&kk[gb];
      *(v8h*)&Kt[kvr][ho] = k8;
      v8h v8 = *(const v8h*)&vv[gb];
#pragma unroll
      for (int e = 0; e < 8; ++e) Vt[ho + e][kvr] = v8[e];
    }
    __syncthreads();

    // S = Q @ K^T
    v8f s[4];
#pragma unroll
    for (int i = 0; i < 4; ++i) s[i] = vzero8();
#pragma unroll
    for (int c = 0; c < 4; ++c) {
#pragma unroll
      for (int nt = 0; nt < 4; ++nt) {
        F16Frag bfr;
        int kvl = nt * 16 + ln;
        int kb  = c * 32 + 8 * hi;
        bfr.h[0] = *(const v8h*)&Kt[kvl][kb];
        bfr.h[1] = *(const v8h*)&Kt[kvl][kb + 16];
        s[nt] = __builtin_amdgcn_wmma_f32_16x16x32_f16(
            false, qf[c].v, false, bfr.v, (short)0, s[nt], false, false);
      }
    }

    // scale + causal mask
    const int rbase = m0 + w * 16 + 8 * hi;
#pragma unroll
    for (int nt = 0; nt < 4; ++nt) {
      int col = j * 64 + nt * 16 + ln;
#pragma unroll
      for (int r = 0; r < 8; ++r) {
        float val = s[nt][r] * scale;
        s[nt][r] = (col > rbase + r) ? -__builtin_inff() : val;
      }
    }

    // online softmax (row stats live in 16-lane halves matching C layout)
    float tmx[8];
#pragma unroll
    for (int r = 0; r < 8; ++r) {
      float m = s[0][r];
#pragma unroll
      for (int nt = 1; nt < 4; ++nt) m = fmaxf(m, s[nt][r]);
      tmx[r] = m;
    }
    for (int off = 1; off < 16; off <<= 1)
#pragma unroll
      for (int r = 0; r < 8; ++r) tmx[r] = fmaxf(tmx[r], __shfl_xor(tmx[r], off, 32));

    float corr[8], tsum[8];
#pragma unroll
    for (int r = 0; r < 8; ++r) {
      float mnew = fmaxf(mrow[r], tmx[r]);
      corr[r] = __expf(mrow[r] - mnew);
      mrow[r] = mnew;
      tsum[r] = 0.0f;
    }
#pragma unroll
    for (int nt = 0; nt < 4; ++nt)
#pragma unroll
      for (int r = 0; r < 8; ++r) {
        float p = __expf(s[nt][r] - mrow[r]);
        tsum[r] += p;
        s[nt][r] = p;
      }
    for (int off = 1; off < 16; off <<= 1)
#pragma unroll
      for (int r = 0; r < 8; ++r) tsum[r] += __shfl_xor(tsum[r], off, 32);
#pragma unroll
    for (int r = 0; r < 8; ++r) lrow[r] = lrow[r] * corr[r] + tsum[r];
#pragma unroll
    for (int nt = 0; nt < 8; ++nt)
#pragma unroll
      for (int r = 0; r < 8; ++r) oacc[nt][r] *= corr[r];

    // restage P (C layout) -> LDS -> A layout
#pragma unroll
    for (int nt = 0; nt < 4; ++nt)
#pragma unroll
      for (int r = 0; r < 8; ++r)
        Pst[w][r + 8 * hi][nt * 16 + ln] = (_Float16)s[nt][r];
    __syncthreads();

    // O += P @ V
#pragma unroll
    for (int c = 0; c < 2; ++c) {
      F16Frag pf;
      int pkb = c * 32 + 8 * hi;
      pf.h[0] = *(const v8h*)&Pst[w][ln][pkb];
      pf.h[1] = *(const v8h*)&Pst[w][ln][pkb + 16];
#pragma unroll
      for (int nt = 0; nt < 8; ++nt) {
        F16Frag bfr;
        int hd = nt * 16 + ln;
        int kb = c * 32 + 8 * hi;
        bfr.h[0] = *(const v8h*)&Vt[hd][kb];
        bfr.h[1] = *(const v8h*)&Vt[hd][kb + 16];
        oacc[nt] = __builtin_amdgcn_wmma_f32_16x16x32_f16(
            false, pf.v, false, bfr.v, (short)0, oacc[nt], false, false);
      }
    }
    __syncthreads();
  }

  float inv[8];
#pragma unroll
  for (int r = 0; r < 8; ++r) inv[r] = 1.0f / lrow[r];
#pragma unroll
  for (int nt = 0; nt < 8; ++nt)
#pragma unroll
    for (int r = 0; r < 8; ++r) {
      int row = m0 + w * 16 + r + 8 * hi;
      int col = h * HDIM + nt * 16 + ln;
      o[((size_t)b * SEQ + row) * DM + col] = (_Float16)(oacc[nt][r] * inv[r]);
    }
}

// ---------------------------------------------------------------------------
extern "C" void kernel_launch(void* const* d_in, const int* in_sizes, int n_in,
                              void* d_out, int out_size, void* d_ws, size_t ws_size,
                              hipStream_t stream) {
  (void)in_sizes; (void)n_in; (void)out_size; (void)ws_size;
  const float* xq = (const float*)d_in[0];
  const float* xk = (const float*)d_in[1];
  const float* xv = (const float*)d_in[2];
  const float* wq = (const float*)d_in[3];
  const float* bq = (const float*)d_in[4];
  const float* wk = (const float*)d_in[5];
  const float* bk = (const float*)d_in[6];
  const float* wv = (const float*)d_in[7];
  const float* bv = (const float*)d_in[8];
  const float* wo = (const float*)d_in[9];
  const float* bo = (const float*)d_in[10];
  float* out = (float*)d_out;

  const int M = BATCH * SEQ;                 // 4096
  const size_t BSD  = (size_t)BATCH * SEQ * DM;
  const size_t BSHD = (size_t)BATCH * SEQ * HDIM;

  char* ws = (char*)d_ws;
  size_t off = 0;
  _Float16* x16  = (_Float16*)(ws + off); off += BSD * 2;   // xq16/xk16/xv16/attn16 (time-shared)
  _Float16* q16  = (_Float16*)(ws + off); off += BSD * 2;
  _Float16* k16  = (_Float16*)(ws + off); off += BSHD * 2;
  _Float16* v16  = (_Float16*)(ws + off); off += BSHD * 2;
  _Float16* wqt  = (_Float16*)(ws + off); off += (size_t)DM * DM * 2;
  _Float16* wkt  = (_Float16*)(ws + off); off += (size_t)DM * HDIM * 2;
  _Float16* wvt  = (_Float16*)(ws + off); off += (size_t)DM * HDIM * 2;
  _Float16* wot  = (_Float16*)(ws + off); off += (size_t)DM * DM * 2;

  dim3 blk(256);

  // one-time weight transpose+convert: W[K,N] f32 -> Wt[N,K] f16
  transpose_cvt_f16<<<dim3(DM / 64, DM / 64), blk, 0, stream>>>(wq, wqt, DM, DM);
  transpose_cvt_f16<<<dim3(HDIM / 64, DM / 64), blk, 0, stream>>>(wk, wkt, DM, HDIM);
  transpose_cvt_f16<<<dim3(HDIM / 64, DM / 64), blk, 0, stream>>>(wv, wvt, DM, HDIM);
  transpose_cvt_f16<<<dim3(DM / 64, DM / 64), blk, 0, stream>>>(wo, wot, DM, DM);

  // Q projection
  cvt_f16<<<dim3((int)(BSD / 2048)), blk, 0, stream>>>(xq, x16, (int)BSD);
  gemm_tn_f16<true><<<dim3(DM / 128, M / 128), blk, 0, stream>>>(
      x16, wqt, bq, (void*)q16, M, DM, DM);
  // K projection
  cvt_f16<<<dim3((int)(BSD / 2048)), blk, 0, stream>>>(xk, x16, (int)BSD);
  gemm_tn_f16<true><<<dim3(HDIM / 128, M / 128), blk, 0, stream>>>(
      x16, wkt, bk, (void*)k16, M, HDIM, DM);
  // V projection
  cvt_f16<<<dim3((int)(BSD / 2048)), blk, 0, stream>>>(xv, x16, (int)BSD);
  gemm_tn_f16<true><<<dim3(HDIM / 128, M / 128), blk, 0, stream>>>(
      x16, wvt, bv, (void*)v16, M, HDIM, DM);
  // causal MQA core -> f16 [B,S,D] into x16 (inputs dead now)
  mqa_flash<<<dim3(SEQ / 128, NH, BATCH), blk, 0, stream>>>(q16, k16, v16, x16);
  // output projection -> f32 d_out
  gemm_tn_f16<false><<<dim3(DM / 128, M / 128), blk, 0, stream>>>(
      x16, wot, bo, (void*)out, M, DM, DM);
}
